// Model_3_10995116278169
// MI455X (gfx1250) — matrix operand order repaired and checked by reading, not compiled, not run on hardware
//
// MI455X / gfx1250 implementation of the hierarchical GNN reference.
//
// Design notes (compile-only reasoning from MI455X specs):
//  * All dense matmuls (GATv2 Wl/Wr, SAGE lin_l/lin_r, head lin1/lin2) use
//    v_wmma_f32_16x16x32_bf16: fp32 activations converted to bf16 fragments
//    in-register; weights pre-packed once into bf16 in B-fragment layout so
//    each lane's 16-element fragment is one contiguous 32B load.
//    One wave32 computes one 16x16 output tile; K and N are template
//    constants so the WMMA loop fully unrolls with immediate-offset loads.
//  * Everything else is memory-bound irregular graph work: wave-per-edge
//    float4 gathers with global f32 atomics for segment max/sum; all live
//    data (< 90 MB) sits in the 192 MB L2, so atomics are L2-resident.
//  * TopK pooling: one workgroup per graph, bitonic sort of <=256 scores in
//    LDS (key = score desc, index asc to match jax.lax.top_k tie behavior).
//
// Input order assumes JAX sorted-key pytree flattening of setup_inputs():
//   0:es 1:et 2:lin1.W 3:lin1.b 4:lin2.W 5:lin2.b 6:lin3.W 7:lin3.b
//   8..25:  s-branch {c1:Wl,Wr,att,b,bl,br  c2:same  c3:Wl,Wr,bl  p1.w p2.w p3.w}
//   26..43: t-branch (same layout)   44:xs  45:xt
//
// Workspace requirement: ~80 MB.

#include <hip/hip_runtime.h>
#include <math.h>

#define Bg   128
#define N0c  256
#define DEGc 16
#define FINc 128
#define Hc   128
#define Nn   (Bg * N0c)          // 32768 nodes
#define Ec   (Bg * N0c * DEGc)   // 524288 edges
#define NEGs 0.01f

typedef float  v8f   __attribute__((ext_vector_type(8)));
typedef __bf16 v16bf __attribute__((ext_vector_type(16)));

__device__ __forceinline__ float leakyf(float x, float s) { return x > 0.f ? x : x * s; }

__device__ __forceinline__ void atomicMaxF(float* addr, float val) {
  int* ia = (int*)addr;
  int old = __float_as_int(*addr);
  while (__int_as_float(old) < val) {
    int assumed = old;
    old = atomicCAS(ia, assumed, __float_as_int(val));
    if (old == assumed) break;
  }
}

// ---------------------------------------------------------------------------
// Weight pre-pack: W[K,N] fp32 (row-major) -> WB bf16 in B-fragment layout:
//   WB[kb*N*32 + n*32 + half*16 + i] = bf16( W[(kb*32 + (i<8 ? 8*half + i : 16 + 8*half + i-8))*N + n] )
// so a lane's 16-element fragment (fixed n=coln, half) is contiguous 32 bytes.
// ---------------------------------------------------------------------------
__global__ void gnn_pack_w(const float* __restrict__ W, __bf16* __restrict__ WB,
                           int K, int N) {
  int i = blockIdx.x * blockDim.x + threadIdx.x;
  if (i >= K * N) return;
  int k = i / N, n = i - (i / N) * N;
  int kb = k >> 5, h = (k >> 3) & 1, e = (k & 7) + (((k >> 4) & 1) << 3);
  WB[(size_t)kb * N * 32 + n * 32 + h * 16 + e] = (__bf16)W[i];
}

// ---------------------------------------------------------------------------
// WMMA GEMM: C[M,N] = act( A[M,K] @ W[K,N] + bias + addend )
// 4 waves per block, one 16x16 output tile per wave. K,N compile-time so the
// WMMA loop fully unrolls with immediate-offset b128 loads.
// A frag (16x32, 16-bit): lane l holds row M=l&15; half h=l>>4;
//   elems 0..7 -> K = k0 + 8h + i ; elems 8..15 -> K = k0 + 16 + 8h + (i-8)
// B frag (32x16): lane l holds col N=l&15; elem i -> same K as A elem i (k0 + 8h + i / k0 + 16 + 8h + (i-8))
// C/D (16x16 f32): lane l col N=l&15; vgpr v row M = v + 8h
// ---------------------------------------------------------------------------
template <int K, int N>
__global__ __launch_bounds__(128) void gnn_gemm(
    const float* __restrict__ A, const __bf16* __restrict__ WB,
    const float* __restrict__ bias, const float* __restrict__ addend,
    float* __restrict__ C, int M, float slope, int act)
{
  int ntiles = (M >> 4) * (N >> 4);
  int tile   = blockIdx.x * 4 + (threadIdx.x >> 5);
  if (tile >= ntiles) return;                       // wave-uniform guard
  constexpr int ntn = N >> 4;
  int tm   = tile / ntn;
  int tn   = tile - tm * ntn;
  int lane = threadIdx.x & 31;
  int half = lane >> 4;
  int coln = tn * 16 + (lane & 15);
  const float*  arow = A + (size_t)(tm * 16 + (lane & 15)) * K + half * 8;
  const __bf16* wrow = WB + (size_t)coln * 32 + half * 16;

  v8f c = {};
#pragma unroll
  for (int kb = 0; kb < K / 32; ++kb) {
    v16bf a, b;
    // A: two 8-float contiguous, 32B-aligned groups -> global_load_b128 pairs
    const float4* ap = (const float4*)(arow + kb * 32);
    float4 a0 = ap[0], a1 = ap[1];                  // K = k0 + 8h + 0..7
    float4 a2 = ap[4], a3 = ap[5];                  // K = k0 + 16 + 8h + 0..7
    a[0]  = (__bf16)a0.x; a[1]  = (__bf16)a0.y; a[2]  = (__bf16)a0.z; a[3]  = (__bf16)a0.w;
    a[4]  = (__bf16)a1.x; a[5]  = (__bf16)a1.y; a[6]  = (__bf16)a1.z; a[7]  = (__bf16)a1.w;
    a[8]  = (__bf16)a2.x; a[9]  = (__bf16)a2.y; a[10] = (__bf16)a2.z; a[11] = (__bf16)a2.w;
    a[12] = (__bf16)a3.x; a[13] = (__bf16)a3.y; a[14] = (__bf16)a3.z; a[15] = (__bf16)a3.w;
    // B: one contiguous 32B fragment (pre-packed bf16), immediate offset
    b = *(const v16bf*)(wrow + (size_t)kb * N * 32);
    c = __builtin_amdgcn_wmma_f32_16x16x32_bf16(false, a, false, b,
                                                (short)0, c, false, false);
  }

  // Branchless epilogue: bias/slope hoisted; addend behind one uniform branch.
  float bv = bias ? bias[coln] : 0.f;
  float sl = act ? slope : 1.f;
  int   base = (tm * 16 + half * 8) * N + coln;
  if (addend) {
#pragma unroll
    for (int v = 0; v < 8; ++v) c[v] += addend[base + v * N];
  }
#pragma unroll
  for (int v = 0; v < 8; ++v) {
    float x = c[v] + bv;
    C[base + v * N] = x > 0.f ? x : x * sl;
  }
}

// ------------------------------ utilities ---------------------------------
__global__ void gnn_fill_f32(float* p, float v, int n) {
  int i = blockIdx.x * blockDim.x + threadIdx.x;
  if (i < n) p[i] = v;
}
__global__ void gnn_fill_i32(int* p, int v, int n) {
  int i = blockIdx.x * blockDim.x + threadIdx.x;
  if (i < n) p[i] = v;
}
__global__ void gnn_init_edges(const int* __restrict__ e, int* es, int* ed, int* em, int ne) {
  int i = blockIdx.x * blockDim.x + threadIdx.x;
  if (i >= ne) return;
  es[i] = e[i];
  ed[i] = e[ne + i];
  em[i] = 1;
}

// --------------------------- GATv2 edge kernels ---------------------------
// items: [0,ne) real edges, [ne, ne+n) self loops. One wave per item.
__global__ void gnn_gat_logit_max(
    const float* __restrict__ xl, const float* __restrict__ xr,
    const float* __restrict__ att,
    const int* __restrict__ es, const int* __restrict__ ed, const int* __restrict__ em,
    float* __restrict__ logit, float* __restrict__ mx, int ne, int n)
{
  int wave = blockIdx.x * (blockDim.x >> 5) + (threadIdx.x >> 5);
  int lane = threadIdx.x & 31;
  if (wave >= ne + n) return;
  int s, d, m;
  if (wave < ne) { s = es[wave]; d = ed[wave]; m = em[wave]; }
  else           { s = wave - ne; d = s; m = 1; }
  float acc = 0.f;
  if (m) {
    float4 l = ((const float4*)(xl + (size_t)s * Hc))[lane];
    float4 r = ((const float4*)(xr + (size_t)d * Hc))[lane];
    float4 a = ((const float4*)att)[lane];
    float t0 = l.x + r.x; t0 = t0 > 0.f ? t0 : 0.2f * t0;
    float t1 = l.y + r.y; t1 = t1 > 0.f ? t1 : 0.2f * t1;
    float t2 = l.z + r.z; t2 = t2 > 0.f ? t2 : 0.2f * t2;
    float t3 = l.w + r.w; t3 = t3 > 0.f ? t3 : 0.2f * t3;
    acc = t0 * a.x + t1 * a.y + t2 * a.z + t3 * a.w;
#pragma unroll
    for (int off = 16; off > 0; off >>= 1) acc += __shfl_xor(acc, off, 32);
  }
  if (lane == 0) {
    if (m) { logit[wave] = acc; atomicMaxF(mx + d, acc); }
    else   { logit[wave] = -__builtin_inff(); }
  }
}

__global__ void gnn_gat_expden(
    const float* __restrict__ logit, const int* __restrict__ ed,
    const float* __restrict__ mx, float* __restrict__ ex, float* __restrict__ den,
    int ne, int n)
{
  int i = blockIdx.x * blockDim.x + threadIdx.x;
  if (i >= ne + n) return;
  int d = (i < ne) ? ed[i] : (i - ne);
  float lg = logit[i];
  float e  = (lg == -__builtin_inff()) ? 0.f : expf(lg - mx[d]);
  ex[i] = e;
  if (e != 0.f) atomicAdd(den + d, e);
}

__global__ void gnn_gat_scatter(
    const float* __restrict__ xl, const float* __restrict__ ex,
    const float* __restrict__ den,
    const int* __restrict__ es, const int* __restrict__ ed,
    float* __restrict__ acc, int ne, int n)
{
  int wave = blockIdx.x * (blockDim.x >> 5) + (threadIdx.x >> 5);
  int lane = threadIdx.x & 31;
  if (wave >= ne + n) return;
  float e = ex[wave];
  if (e == 0.f) return;
  int s, d;
  if (wave < ne) { s = es[wave]; d = ed[wave]; }
  else           { s = wave - ne; d = s; }
  float alpha = e / den[d];
  float4 v = ((const float4*)(xl + (size_t)s * Hc))[lane];
  float* o = acc + (size_t)d * Hc + lane * 4;
  atomicAdd(o + 0, alpha * v.x);
  atomicAdd(o + 1, alpha * v.y);
  atomicAdd(o + 2, alpha * v.z);
  atomicAdd(o + 3, alpha * v.w);
}

__global__ void gnn_bias_act(float* a, const float* __restrict__ b, int n, float slope) {
  int i = blockIdx.x * blockDim.x + threadIdx.x;
  if (i >= n * Hc) return;
  a[i] = leakyf(a[i] + b[i & (Hc - 1)], slope);
}

// ------------------------------ TopK pooling ------------------------------
// One block per graph; bitonic sort (score desc, idx asc) of npg elements.
__global__ void gnn_topk_pool(
    const float* __restrict__ xin, const float* __restrict__ w,
    float* __restrict__ xout, int* __restrict__ newid, int* __restrict__ km, int npg)
{
  __shared__ float sc[256];
  __shared__ int   si[256];
  __shared__ float swn;
  int g = blockIdx.x, t = threadIdx.x;
  if (t == 0) {
    float s = 0.f;
    for (int j = 0; j < Hc; ++j) s += w[j] * w[j];
    swn = sqrtf(s);
  }
  __syncthreads();
  if (t < npg) {
    int node = g * npg + t;
    float dot = 0.f;
    for (int j = 0; j < Hc; ++j) dot += xin[(size_t)node * Hc + j] * w[j];
    sc[t] = tanhf(dot / swn);
    si[t] = t;
  }
  __syncthreads();
  for (int ksz = 2; ksz <= npg; ksz <<= 1) {
    for (int j = ksz >> 1; j > 0; j >>= 1) {
      if (t < npg) {
        int ixj = t ^ j;
        if (ixj > t) {
          float a = sc[t], b = sc[ixj];
          int   ia = si[t], ib = si[ixj];
          bool first   = (a > b) || (a == b && ia < ib);   // (a,ia) ranks earlier
          bool dirDesc = ((t & ksz) == 0);
          if (dirDesc ? !first : first) {
            sc[t] = b;  sc[ixj] = a;
            si[t] = ib; si[ixj] = ia;
          }
        }
      }
      __syncthreads();
    }
  }
  int k = npg >> 1;
  if (t < k) {
    int old = g * npg + si[t];
    int nw  = g * k + t;
    float sv = sc[t];
    for (int j = 0; j < Hc; ++j)
      xout[(size_t)nw * Hc + j] = xin[(size_t)old * Hc + j] * sv;
    newid[old] = nw;
    km[old] = 1;
  }
}

__global__ void gnn_edge_remap(int* es, int* ed, int* em,
                               const int* __restrict__ newid, const int* __restrict__ km, int ne) {
  int e = blockIdx.x * blockDim.x + threadIdx.x;
  if (e >= ne) return;
  int s = es[e], d = ed[e];
  em[e] = (em[e] && km[s] && km[d]) ? 1 : 0;
  es[e] = newid[s];
  ed[e] = newid[d];
}

__global__ void gnn_readout_add(const float* __restrict__ x, float* rd, int npg) {
  int g = blockIdx.x, f = threadIdx.x;  // blockDim = Hc
  float mx = -__builtin_inff(), sm = 0.f;
  for (int i = 0; i < npg; ++i) {
    float v = x[(size_t)(g * npg + i) * Hc + f];
    mx = fmaxf(mx, v);
    sm += v;
  }
  rd[g * 2 * Hc + f]      += mx;
  rd[g * 2 * Hc + Hc + f] += sm / (float)npg;
}

// ------------------------------- SAGE conv --------------------------------
__global__ void gnn_sage_scatter(
    const float* __restrict__ x, const int* __restrict__ es, const int* __restrict__ ed,
    const int* __restrict__ em, float* __restrict__ ssum, float* __restrict__ cnt, int ne)
{
  int wave = blockIdx.x * (blockDim.x >> 5) + (threadIdx.x >> 5);
  int lane = threadIdx.x & 31;
  if (wave >= ne) return;
  if (!em[wave]) return;
  int s = es[wave], d = ed[wave];
  float4 v = ((const float4*)(x + (size_t)s * Hc))[lane];
  float* o = ssum + (size_t)d * Hc + lane * 4;
  atomicAdd(o + 0, v.x);
  atomicAdd(o + 1, v.y);
  atomicAdd(o + 2, v.z);
  atomicAdd(o + 3, v.w);
  if (lane == 0) atomicAdd(cnt + d, 1.0f);
}

__global__ void gnn_sage_mean(float* ssum, const float* __restrict__ cnt, int n) {
  int i = blockIdx.x * blockDim.x + threadIdx.x;
  if (i >= n * Hc) return;
  ssum[i] /= fmaxf(cnt[i >> 7], 1.0f);
}

// -------------------------------- head ------------------------------------
__global__ void gnn_head_concat(const float* __restrict__ rs, const float* __restrict__ rt,
                                float* hb) {
  int i = blockIdx.x * blockDim.x + threadIdx.x;
  if (i >= Bg * 512) return;
  int g = i >> 9, c = i & 511;
  hb[i] = (c < 256) ? rs[g * 256 + c] : rt[g * 256 + (c - 256)];
}

__global__ void gnn_head_final(const float* __restrict__ h2, const float* __restrict__ W3,
                               const float* __restrict__ b3, float* out) {
  int g = blockIdx.x * blockDim.x + threadIdx.x;
  if (g >= Bg) return;
  float l0 = b3[0], l1 = b3[1];
  for (int j = 0; j < 64; ++j) {
    float v = h2[g * 64 + j];
    l0 += v * W3[j * 2 + 0];
    l1 += v * W3[j * 2 + 1];
  }
  float m   = fmaxf(l0, l1);
  float lse = m + logf(expf(l0 - m) + expf(l1 - m));
  out[g * 2 + 0] = l0 - lse;
  out[g * 2 + 1] = l1 - lse;
}

// ------------------------------ host driver -------------------------------
static inline int cdiv(int a, int b) { return (a + b - 1) / b; }

extern "C" void kernel_launch(void* const* d_in, const int* in_sizes, int n_in,
                              void* d_out, int out_size, void* d_ws, size_t ws_size,
                              hipStream_t stream) {
  (void)in_sizes; (void)n_in; (void)out_size; (void)ws_size;

  // ---- workspace layout (bump allocator) ----
  float* X     = (float*)d_ws;              // Nn*Hc
  float* XL    = X   + (size_t)Nn * Hc;
  float* XR    = XL  + (size_t)Nn * Hc;
  float* ACC   = XR  + (size_t)Nn * Hc;
  float* LOGIT = ACC + (size_t)Nn * Hc;     // Ec+Nn
  float* EXb   = LOGIT + (Ec + Nn);
  float* MX    = EXb + (Ec + Nn);           // Nn
  float* DEN   = MX + Nn;                   // Nn (reused as SAGE cnt)
  int*   ES    = (int*)(DEN + Nn);          // Ec
  int*   ED    = ES + Ec;
  int*   EM    = ED + Ec;
  int*   NEWID = EM + Ec;                   // Nn
  int*   KM    = NEWID + Nn;                // Nn
  float* RDS   = (float*)(KM + Nn);         // Bg*2*Hc
  float* RDT   = RDS + Bg * 2 * Hc;
  float* HB    = RDT + Bg * 2 * Hc;         // Bg*512
  float* H1    = HB + Bg * 512;             // Bg*Hc
  float* H2    = H1 + Bg * Hc;              // Bg*64
  __bf16* WB   = (__bf16*)(H2 + Bg * 64);   // up to 512*128 bf16 (packed weights)

  const float NINF = -__builtin_inff();

  auto gemm = [&](const float* A, const float* W, const float* bias, const float* add,
                  float* C, int M, int K, int N, float slope, int act) {
    gnn_pack_w<<<cdiv(K * N, 256), 256, 0, stream>>>(W, WB, K, N);
    dim3 grid(cdiv((M / 16) * (N / 16), 4));
    if (K == 512 && N == 128)
      gnn_gemm<512, 128><<<grid, 128, 0, stream>>>(A, WB, bias, add, C, M, slope, act);
    else if (K == 128 && N == 64)
      gnn_gemm<128, 64><<<grid, 128, 0, stream>>>(A, WB, bias, add, C, M, slope, act);
    else
      gnn_gemm<128, 128><<<grid, 128, 0, stream>>>(A, WB, bias, add, C, M, slope, act);
  };
  auto fillf = [&](float* p, float v, int n) {
    gnn_fill_f32<<<cdiv(n, 256), 256, 0, stream>>>(p, v, n);
  };
  auto filli = [&](int* p, int v, int n) {
    gnn_fill_i32<<<cdiv(n, 256), 256, 0, stream>>>(p, v, n);
  };

  // GATv2 conv: din->Hc on n nodes; result (post bias + leaky NEGs) in ACC.
  auto gat = [&](const float* xin, int din, int n, const float* Wl, const float* Wr,
                 const float* att, const float* bconv, const float* bl, const float* br) {
    gemm(xin, Wl, bl, nullptr, XL, n, din, Hc, 0.f, 0);
    gemm(xin, Wr, br, nullptr, XR, n, din, Hc, 0.f, 0);
    fillf(MX, NINF, n);
    fillf(DEN, 0.f, n);
    fillf(ACC, 0.f, n * Hc);
    int items = Ec + n;
    gnn_gat_logit_max<<<cdiv(items, 8), 256, 0, stream>>>(XL, XR, att, ES, ED, EM, LOGIT,
                                                          MX, Ec, n);
    gnn_gat_expden<<<cdiv(items, 256), 256, 0, stream>>>(LOGIT, ED, MX, EXb, DEN, Ec, n);
    gnn_gat_scatter<<<cdiv(items, 8), 256, 0, stream>>>(XL, EXb, DEN, ES, ED, ACC, Ec, n);
    gnn_bias_act<<<cdiv(n * Hc, 256), 256, 0, stream>>>(ACC, bconv, n, NEGs);
  };

  // TopK pool from ACC (npg -> npg/2), write X, remap edges, accumulate readout.
  auto pool = [&](int npg, const float* w, float* rd, bool remap) {
    int n = Bg * npg;
    filli(NEWID, 0, n);
    filli(KM, 0, n);
    gnn_topk_pool<<<Bg, 256, 0, stream>>>(ACC, w, X, NEWID, KM, npg);
    if (remap) gnn_edge_remap<<<cdiv(Ec, 256), 256, 0, stream>>>(ES, ED, EM, NEWID, KM, Ec);
    gnn_readout_add<<<Bg, Hc, 0, stream>>>(X, rd, npg >> 1);
  };

  // Branch param layout (sorted-key flatten):
  //  0:c1.Wl 1:c1.Wr 2:c1.att 3:c1.b 4:c1.bl 5:c1.br
  //  6..11: c2 (same)  12:c3.Wl 13:c3.Wr 14:c3.bl  15:p1.w 16:p2.w 17:p3.w
  auto branch = [&](const float* x0, const int* e_in, const float* const* P, float* rd) {
    gnn_init_edges<<<cdiv(Ec, 256), 256, 0, stream>>>(e_in, ES, ED, EM, Ec);
    fillf(rd, 0.f, Bg * 2 * Hc);
    // conv1 (GATv2, FIN->H) + pool1 (256->128)
    gat(x0, FINc, Nn, P[0], P[1], P[2], P[3], P[4], P[5]);
    pool(256, P[15], rd, true);
    // conv2 (GATv2, H->H, n=16384) + pool2 (128->64)
    gat(X, Hc, Bg * 128, P[6], P[7], P[8], P[9], P[10], P[11]);
    pool(128, P[16], rd, true);
    // conv3 (SAGE, n=8192): out = mean@Wl + bl + x@Wr, then leaky NEGs
    int n3 = Bg * 64;
    gemm(X, P[13], nullptr, nullptr, XL, n3, Hc, Hc, 0.f, 0);   // x @ Wr
    fillf(XR, 0.f, n3 * Hc);
    fillf(DEN, 0.f, n3);
    gnn_sage_scatter<<<cdiv(Ec, 8), 256, 0, stream>>>(X, ES, ED, EM, XR, DEN, Ec);
    gnn_sage_mean<<<cdiv(n3 * Hc, 256), 256, 0, stream>>>(XR, DEN, n3);
    gemm(XR, P[12], P[14], XL, ACC, n3, Hc, Hc, NEGs, 1);       // + leaky
    // pool3 (64->32), no edge remap needed afterwards
    pool(64, P[17], rd, false);
  };

  const float* Ps[18];
  const float* Pt[18];
  for (int i = 0; i < 18; ++i) {
    Ps[i] = (const float*)d_in[8 + i];
    Pt[i] = (const float*)d_in[26 + i];
  }

  branch((const float*)d_in[44], (const int*)d_in[0], Ps, RDS);  // source branch
  branch((const float*)d_in[45], (const int*)d_in[1], Pt, RDT);  // target branch

  // Head: concat -> lin1(512->128)+leaky -> lin2(128->64)+leaky -> lin3 + log_softmax
  gnn_head_concat<<<cdiv(Bg * 512, 256), 256, 0, stream>>>(RDS, RDT, HB);
  gemm(HB, (const float*)d_in[2], (const float*)d_in[3], nullptr, H1, Bg, 512, Hc, NEGs, 1);
  gemm(H1, (const float*)d_in[4], (const float*)d_in[5], nullptr, H2, Bg, Hc, 64, NEGs, 1);
  gnn_head_final<<<1, Bg, 0, stream>>>(H2, (const float*)d_in[6], (const float*)d_in[7],
                                       (float*)d_out);
}
